// INT4Linear_46076409152226
// MI455X (gfx1250) — compile-verified, matrix operand order checked by reading, not run on hardware
//
#include <hip/hip_runtime.h>

typedef __attribute__((ext_vector_type(4)))  float        f32x4;
typedef __attribute__((ext_vector_type(8)))  float        f32x8;
typedef __attribute__((ext_vector_type(4)))  int          i32x4;
typedef __attribute__((ext_vector_type(8)))  int          i32x8;
typedef __attribute__((ext_vector_type(4)))  unsigned int u32x4;
typedef __attribute__((ext_vector_type(4)))  __bf16       bf16x4;
typedef __attribute__((ext_vector_type(8)))  __bf16       bf16x8;
typedef __attribute__((ext_vector_type(16))) __bf16       bf16x16;

#define BM 128
#define BN 256
#define BK 32
#define KPAD (BK + 8)   // 80B LDS row stride (64B data + 16B pad)

// ---------------------------------------------------------------------------
// Pass 1: one-time int4-code -> bf16 weight dequant into workspace.
// ---------------------------------------------------------------------------
__global__ __launch_bounds__(256)
void dequant_w_kernel(const int* __restrict__ Wq, const float* __restrict__ scale_p,
                      __bf16* __restrict__ Wb)
{
    const float scale = scale_p[0];
    const long long i = ((long long)blockIdx.x * 256 + threadIdx.x) * 8;
    #pragma unroll
    for (int h = 0; h < 2; ++h) {
        i32x4 q = *(const i32x4*)(Wq + i + h * 4);
        bf16x4 t = { (__bf16)((float)(q.x - 8) * scale),
                     (__bf16)((float)(q.y - 8) * scale),
                     (__bf16)((float)(q.z - 8) * scale),
                     (__bf16)((float)(q.w - 8) * scale) };
        *(bf16x4*)(Wb + i + h * 4) = t;
    }
}

// ---------------------------------------------------------------------------
// TDM descriptor issue: 2D tile (BN rows x BK bf16) from row-major [N][K]
// bf16 matrix into LDS at lds_off, rows padded 64B->80B (pad 4 DW / 16 DW).
// D# fields per CDNA5 ISA ch.8 (group0/group1 bit layout).
// amdgpu-toolchain (clang-23) 6-arg builtin form.
// ---------------------------------------------------------------------------
__device__ __forceinline__ void tdm_load_tile_b(unsigned long long gaddr,
                                                unsigned lds_off, int K, int N)
{
    u32x4 g0;
    g0[0] = 1u;                                           // count=1, user descriptor
    g0[1] = lds_off;                                      // lds_addr (bytes)
    g0[2] = (unsigned)(gaddr & 0xFFFFFFFFull);            // global_addr[31:0]
    g0[3] = (unsigned)((gaddr >> 32) & 0x01FFFFFFull)     // global_addr[56:32]
          | (2u << 30);                                   // type=2 ("image")
    i32x8 g1;
    g1[0] = (1 << 16)                                     // data_size = 2 bytes
          | (1 << 20)                                     // pad_enable
          | (3 << 22)                                     // pad_interval: 16 DWORDs (64B)
          | (3 << 25);                                    // pad_amount: 4 DWORDs (16B)
    g1[1] = (int)(((unsigned)K & 0xFFFFu) << 16);         // tensor_dim0[15:0] (K elems)
    g1[2] = (int)((((unsigned)K >> 16) & 0xFFFFu)         // tensor_dim0[31:16]
          | (((unsigned)N & 0xFFFFu) << 16));             // tensor_dim1[15:0]
    g1[3] = (int)((((unsigned)N >> 16) & 0xFFFFu)         // tensor_dim1[31:16]
          | ((unsigned)BK << 16));                        // tile_dim0 = 32
    g1[4] = BN;                                           // tile_dim1 = 256 (tile_dim2=0)
    g1[5] = K;                                            // tensor_dim0_stride[31:0]
    g1[6] = 0;                                            // stride hi / dim1_stride lo
    g1[7] = 0;
    i32x4 z4 = {0, 0, 0, 0};
    i32x8 z8 = {0, 0, 0, 0, 0, 0, 0, 0};
    __builtin_amdgcn_tensor_load_to_lds(g0, g1, z4, z4, z8, 0);
}

// ---------------------------------------------------------------------------
// Pass 2: GEMM. A: f32 global -> cvt -> LDS (double buffer, reg prefetch).
//         B: bf16 workspace -> LDS via TDM (triple buffer, TENSORcnt pipeline).
// ---------------------------------------------------------------------------
__global__ __launch_bounds__(256)
void int4_linear_wmma_tdm(const float* __restrict__ X,
                          const __bf16* __restrict__ Wb,
                          const float* __restrict__ bias,
                          float* __restrict__ Y,
                          int M, int N, int K)
{
    __shared__ __align__(16) __bf16 As[2][BM][KPAD];
    __shared__ __align__(16) __bf16 Bs[3][BN][KPAD];

    const int tid  = threadIdx.x;
    const int lane = tid & 31;
    const int wave = tid >> 5;
    const int wm   = wave & 1;
    const int wn   = wave >> 1;

    const int blockN = blockIdx.x * BN;
    const int blockM = blockIdx.y * BM;

    const int lrow = tid >> 3;
    const int lcol = (tid & 7) * 4;

    const int frow  = lane & 15;
    const int khalf = (lane >> 4) * 8;

    const int T = K / BK;
    const unsigned long long bRowBase =
        (unsigned long long)(size_t)(Wb + (size_t)blockN * K);
    const unsigned bsLds0 = (unsigned)(size_t)&Bs[0][0][0];
    const unsigned bsStride = BN * KPAD * 2;   // bytes per B buffer

    f32x8 acc[4][4] = {};

    // prologue: stage A tile 0; issue TDM for B tiles 0 and 1
    #pragma unroll
    for (int p = 0; p < BM / 32; ++p) {
        const int m = p * 32 + lrow;
        f32x4 v = *(const f32x4*)(X + (size_t)(blockM + m) * K + lcol);
        bf16x4 t = { (__bf16)v.x, (__bf16)v.y, (__bf16)v.z, (__bf16)v.w };
        *(bf16x4*)&As[0][m][lcol] = t;
    }
    if (wave == 0) {
        tdm_load_tile_b(bRowBase, bsLds0, K, N);
        if (T > 1) tdm_load_tile_b(bRowBase + (size_t)BK * 2, bsLds0 + bsStride, K, N);
    }

    int bufB = 0;
    for (int t = 0; t < T; ++t) {
        const int  k0       = t * BK;
        const bool has_next = (t + 1) < T;
        const int  curA     = t & 1;
        const int  nxtA     = curA ^ 1;

        // tile t resident (tile t+1 may still be in flight)
        if (wave == 0) {
            if (has_next) __builtin_amdgcn_s_wait_tensorcnt(1);
            else          __builtin_amdgcn_s_wait_tensorcnt(0);
        }
        __syncthreads();   // releases B tile t and A parks from iteration t-1

        // DMA tile t+2 into buffer (t+2)%3: its previous readers finished at
        // iteration t-1, which the barrier above has already ordered.
        if (wave == 0 && (t + 2) < T) {
            const int b2 = (bufB + 2 >= 3) ? (bufB - 1) : (bufB + 2);
            tdm_load_tile_b(bRowBase + (size_t)(k0 + 2 * BK) * 2,
                            bsLds0 + (unsigned)b2 * bsStride, K, N);
        }

        // A prefetch for tile t+1 (in flight under the WMMAs)
        f32x4 av[BM / 32];
        if (has_next) {
            const int kn = k0 + BK;
            #pragma unroll
            for (int p = 0; p < BM / 32; ++p)
                av[p] = *(const f32x4*)(X + (size_t)(blockM + p * 32 + lrow) * K + kn + lcol);
        }

        // 4x4 WMMA block (64x64 wave tile)
        bf16x16 bfrag[4];
        #pragma unroll
        for (int nf = 0; nf < 4; ++nf) {
            const int n = wn * 64 + nf * 16 + frow;
            bf16x8 lo = *(const bf16x8*)&Bs[bufB][n][khalf];
            bf16x8 hi = *(const bf16x8*)&Bs[bufB][n][16 + khalf];
            bfrag[nf] = __builtin_shufflevector(lo, hi,
                0,1,2,3,4,5,6,7,8,9,10,11,12,13,14,15);
        }
        #pragma unroll
        for (int mf = 0; mf < 4; ++mf) {
            const int m = wm * 64 + mf * 16 + frow;
            bf16x8 lo = *(const bf16x8*)&As[curA][m][khalf];
            bf16x8 hi = *(const bf16x8*)&As[curA][m][16 + khalf];
            bf16x16 afrag = __builtin_shufflevector(lo, hi,
                0,1,2,3,4,5,6,7,8,9,10,11,12,13,14,15);
            #pragma unroll
            for (int nf = 0; nf < 4; ++nf) {
                acc[mf][nf] = __builtin_amdgcn_wmma_f32_16x16x32_bf16(
                    false, afrag, false, bfrag[nf],
                    (short)0, acc[mf][nf], false, false);
            }
        }

        // park A tile t+1 (visibility covered by next iteration's barrier)
        if (has_next) {
            #pragma unroll
            for (int p = 0; p < BM / 32; ++p) {
                bf16x4 ta = { (__bf16)av[p].x, (__bf16)av[p].y,
                              (__bf16)av[p].z, (__bf16)av[p].w };
                *(bf16x4*)&As[nxtA][p * 32 + lrow][lcol] = ta;
            }
        }

        bufB = (bufB + 1 == 3) ? 0 : bufB + 1;
    }

    // epilogue: fused bias add
    const int nlane = lane & 15;
    const int mhalf = (lane >> 4) * 8;
    #pragma unroll
    for (int nf = 0; nf < 4; ++nf) {
        const int gcol = blockN + wn * 64 + nf * 16 + nlane;
        const float b = bias[gcol];
        #pragma unroll
        for (int mf = 0; mf < 4; ++mf) {
            const int gmbase = blockM + wm * 64 + mf * 16 + mhalf;
            #pragma unroll
            for (int r = 0; r < 8; ++r)
                Y[(size_t)(gmbase + r) * N + gcol] = acc[mf][nf][r] + b;
        }
    }
}

// ---------------------------------------------------------------------------
// Fallback: fully fused single-kernel path (round-2 version),
// used when the workspace cannot hold the bf16 weight matrix.
// ---------------------------------------------------------------------------
__global__ __launch_bounds__(256)
void int4_linear_wmma_fused(const float* __restrict__ X,
                            const int*   __restrict__ Wq,
                            const float* __restrict__ scale_p,
                            const float* __restrict__ bias,
                            float* __restrict__ Y,
                            int M, int N, int K)
{
    __shared__ __align__(16) __bf16 As[2][BM][KPAD];
    __shared__ __align__(16) __bf16 Bs[2][BN][KPAD];

    const int tid  = threadIdx.x;
    const int lane = tid & 31;
    const int wave = tid >> 5;
    const int wm   = wave & 1;
    const int wn   = wave >> 1;
    const int blockN = blockIdx.x * BN;
    const int blockM = blockIdx.y * BM;
    const float scale = scale_p[0];
    const int lrow = tid >> 3;
    const int lcol = (tid & 7) * 4;
    const int frow  = lane & 15;
    const int khalf = (lane >> 4) * 8;

    f32x8 acc[4][4] = {};

    #pragma unroll
    for (int p = 0; p < BM / 32; ++p) {
        const int m = p * 32 + lrow;
        f32x4 v = *(const f32x4*)(X + (size_t)(blockM + m) * K + lcol);
        bf16x4 t = { (__bf16)v.x, (__bf16)v.y, (__bf16)v.z, (__bf16)v.w };
        *(bf16x4*)&As[0][m][lcol] = t;
    }
    #pragma unroll
    for (int p = 0; p < BN / 32; ++p) {
        const int n = p * 32 + lrow;
        i32x4 q = *(const i32x4*)(Wq + (size_t)(blockN + n) * K + lcol);
        bf16x4 t = { (__bf16)((float)(q.x - 8) * scale),
                     (__bf16)((float)(q.y - 8) * scale),
                     (__bf16)((float)(q.z - 8) * scale),
                     (__bf16)((float)(q.w - 8) * scale) };
        *(bf16x4*)&Bs[0][n][lcol] = t;
    }
    __syncthreads();

    for (int k0 = 0; k0 < K; k0 += BK) {
        const int  cur      = (k0 / BK) & 1;
        const int  nxt      = cur ^ 1;
        const bool has_next = (k0 + BK) < K;

        f32x4 av[BM / 32];
        i32x4 bv[BN / 32];
        if (has_next) {
            const int kn = k0 + BK;
            #pragma unroll
            for (int p = 0; p < BM / 32; ++p)
                av[p] = *(const f32x4*)(X + (size_t)(blockM + p * 32 + lrow) * K + kn + lcol);
            #pragma unroll
            for (int p = 0; p < BN / 32; ++p)
                bv[p] = *(const i32x4*)(Wq + (size_t)(blockN + p * 32 + lrow) * K + kn + lcol);
        }

        bf16x16 bfrag[4];
        #pragma unroll
        for (int nf = 0; nf < 4; ++nf) {
            const int n = wn * 64 + nf * 16 + frow;
            bf16x8 lo = *(const bf16x8*)&Bs[cur][n][khalf];
            bf16x8 hi = *(const bf16x8*)&Bs[cur][n][16 + khalf];
            bfrag[nf] = __builtin_shufflevector(lo, hi,
                0,1,2,3,4,5,6,7,8,9,10,11,12,13,14,15);
        }
        #pragma unroll
        for (int mf = 0; mf < 4; ++mf) {
            const int m = wm * 64 + mf * 16 + frow;
            bf16x8 lo = *(const bf16x8*)&As[cur][m][khalf];
            bf16x8 hi = *(const bf16x8*)&As[cur][m][16 + khalf];
            bf16x16 afrag = __builtin_shufflevector(lo, hi,
                0,1,2,3,4,5,6,7,8,9,10,11,12,13,14,15);
            #pragma unroll
            for (int nf = 0; nf < 4; ++nf) {
                acc[mf][nf] = __builtin_amdgcn_wmma_f32_16x16x32_bf16(
                    false, afrag, false, bfrag[nf],
                    (short)0, acc[mf][nf], false, false);
            }
        }

        if (has_next) {
            #pragma unroll
            for (int p = 0; p < BM / 32; ++p) {
                bf16x4 t = { (__bf16)av[p].x, (__bf16)av[p].y,
                             (__bf16)av[p].z, (__bf16)av[p].w };
                *(bf16x4*)&As[nxt][p * 32 + lrow][lcol] = t;
            }
            #pragma unroll
            for (int p = 0; p < BN / 32; ++p) {
                bf16x4 t = { (__bf16)((float)(bv[p].x - 8) * scale),
                             (__bf16)((float)(bv[p].y - 8) * scale),
                             (__bf16)((float)(bv[p].z - 8) * scale),
                             (__bf16)((float)(bv[p].w - 8) * scale) };
                *(bf16x4*)&Bs[nxt][p * 32 + lrow][lcol] = t;
            }
        }
        __syncthreads();
    }

    const int nlane = lane & 15;
    const int mhalf = (lane >> 4) * 8;
    #pragma unroll
    for (int nf = 0; nf < 4; ++nf) {
        const int gcol = blockN + wn * 64 + nf * 16 + nlane;
        const float b = bias[gcol];
        #pragma unroll
        for (int mf = 0; mf < 4; ++mf) {
            const int gmbase = blockM + wm * 64 + mf * 16 + mhalf;
            #pragma unroll
            for (int r = 0; r < 8; ++r)
                Y[(size_t)(gmbase + r) * N + gcol] = acc[mf][nf][r] + b;
        }
    }
}

extern "C" void kernel_launch(void* const* d_in, const int* in_sizes, int n_in,
                              void* d_out, int out_size, void* d_ws, size_t ws_size,
                              hipStream_t stream) {
    const float* x     = (const float*)d_in[0];
    const int*   wq    = (const int*)  d_in[1];
    const float* scale = (const float*)d_in[2];
    const float* bias  = (const float*)d_in[3];
    float*       y     = (float*)d_out;

    const int       OUT = in_sizes[3];                 // 4096
    const long long IN  = in_sizes[1] / OUT;           // 4096
    const long long M   = in_sizes[0] / IN;            // 8192

    const size_t wbytes = (size_t)OUT * (size_t)IN * sizeof(__bf16);
    dim3 grid(OUT / BN, (int)(M / BM));

    if (ws_size >= wbytes) {
        __bf16* wb = (__bf16*)d_ws;
        const long long welems = (long long)OUT * IN;
        dequant_w_kernel<<<(int)(welems / 2048), 256, 0, stream>>>(wq, scale, wb);
        int4_linear_wmma_tdm<<<grid, 256, 0, stream>>>(x, wb, bias, y,
                                                       (int)M, OUT, (int)IN);
    } else {
        int4_linear_wmma_fused<<<grid, 256, 0, stream>>>(x, wq, scale, bias, y,
                                                         (int)M, OUT, (int)IN);
    }
}
